// G_Tensor3D_82738249990463
// MI455X (gfx1250) — matrix operand order repaired and checked by reading, not compiled, not run on hardware
//
#include <hip/hip_runtime.h>

// ---------------------------------------------------------------------------
// Fused bilinear-featurize + z-modulation + MLP(32->32->32->1) for gfx1250.
// All GEMM work runs on v_wmma_f32_16x16x32_f16 (f16 in, f32 accumulate).
// ---------------------------------------------------------------------------

typedef __attribute__((ext_vector_type(16))) _Float16 v16h;
typedef __attribute__((ext_vector_type(8)))  float    v8f;

#define MODE    512
#define NF      32
#define ZD      64
#define NZ      8
#define NPIX    (512 * 512)
#define HSTRIDE 36            // padded f32 row stride for LDS staging (16B-aligned, conflict-free)

__device__ __forceinline__ v8f wmma_f16(v16h a, v16h b, v8f c) {
  // (neg_a, A, neg_b, B, c_mod, C, reuse_a, reuse_b)
  return __builtin_amdgcn_wmma_f32_16x16x32_f16(false, a, false, b, (short)0, c,
                                                false, false);
}

__device__ __forceinline__ void lds_fence() {
  // Within-wave LDS ops are in-order; this pins compiler ordering and waits
  // out DScnt so cross-lane store->load through LDS is safe without a
  // workgroup barrier.
  asm volatile("s_wait_dscnt 0" ::: "memory");
}

__global__ __launch_bounds__(256, 2)
void fused_grid_mlp(const float* __restrict__ z,
                    const float* __restrict__ data,
                    const float* __restrict__ z_data,
                    const float* __restrict__ lerp,
                    const int*   __restrict__ x0a,
                    const int*   __restrict__ y0a,
                    const int*   __restrict__ x1a,
                    const int*   __restrict__ y1a,
                    const float* __restrict__ W1, const float* __restrict__ b1,
                    const float* __restrict__ W2, const float* __restrict__ b2,
                    const float* __restrict__ W3, const float* __restrict__ b3,
                    float* __restrict__ out)
{
  __shared__ float zfeat[NZ * NF];            // 8 x 32 z-lerped features
  __shared__ float hstage[8][16 * HSTRIDE];   // per-wave D->A transpose staging

  const int tid = threadIdx.x;

  // ---- z-feature table, once per block (256 threads == 8*32 entries) ------
  {
    const int zi = tid >> 5;
    const int k  = tid & 31;
    const float zv = z[zi];
    const float zn = (float)(ZD - 1) * zv;     // Z_MIN=0, Z_MAX=1
    const float zt = truncf(zn);
    int z0 = (int)zt;
    z0 = z0 < 0 ? 0 : (z0 > ZD - 1 ? ZD - 1 : z0);
    const int z1 = (z0 + 1 > ZD - 1) ? (ZD - 1) : (z0 + 1);
    const float zw = zn - zt;
    zfeat[zi * NF + k] = z_data[z0 * NF + k] * (1.0f - zw)
                       + z_data[z1 * NF + k] * zw;
  }
  __syncthreads();

  const int lane = tid & 31;
  const int wv   = tid >> 5;
  const int half = lane >> 4;    // which K-half of the A operand this lane holds
  const int mrow = lane & 15;    // A-operand row / B-operand column (local)
  const int kb   = 8 * half;     // first K-run base: {0..7,16..23} vs {8..15,24..31}

  float* hbuf = hstage[wv];

  // ---- weights into WMMA B-operand layout (once per wave) -----------------
  // B 32x16 layout: lane = column n (per 16-lane half), K = (e<8) ? kb+e : 16+kb+(e-8).
  const int kw = kb;
  v16h B1m[2], B2m[2], B3m;
#pragma unroll
  for (int h = 0; h < 2; ++h) {
    const int n = 16 * h + mrow;
#pragma unroll
    for (int e = 0; e < 16; ++e) {
      B1m[h][e] = (_Float16)W1[(kw + (e < 8 ? e : 8 + e)) * 32 + n];
      B2m[h][e] = (_Float16)W2[(kw + (e < 8 ? e : 8 + e)) * 32 + n];
    }
  }
#pragma unroll
  for (int e = 0; e < 16; ++e)
    B3m[e] = (mrow == 0) ? (_Float16)W3[kw + (e < 8 ? e : 8 + e)] : (_Float16)0.0f;

  const float bias1[2] = { b1[mrow], b1[16 + mrow] };
  const float bias2[2] = { b2[mrow], b2[16 + mrow] };
  const float bias3    = b3[0];

  // ---- bilinear gather for this wave's 16-pixel tile ----------------------
  const int tile = (blockIdx.x * 8 + wv) * 16;
  const int p    = tile + mrow;

  const float wx = lerp[2 * p + 0];
  const float wy = lerp[2 * p + 1];
  const int xx0 = x0a[p], yy0 = y0a[p], xx1 = x1a[p], yy1 = y1a[p];
  const float w00 = (1.0f - wx) * (1.0f - wy);
  const float w01 = wx * (1.0f - wy);
  const float w10 = (1.0f - wx) * wy;
  const float w11 = wx * wy;

  const float* c00 = data + ((size_t)yy0 * MODE + xx0) * NF;
  const float* c01 = data + ((size_t)yy0 * MODE + xx1) * NF;
  const float* c10 = data + ((size_t)yy1 * MODE + xx0) * NF;
  const float* c11 = data + ((size_t)yy1 * MODE + xx1) * NF;

  // xy[0..7]  = features kb..kb+7 ; xy[8..15] = features 16+kb..16+kb+7
  float xy[16];
#pragma unroll
  for (int j = 0; j < 4; ++j) {
    const int o = (j < 2) ? (kb + 4 * j) : (16 + kb + 4 * (j - 2));
    const float4 a = *(const float4*)(c00 + o);
    const float4 b = *(const float4*)(c01 + o);
    const float4 c = *(const float4*)(c10 + o);
    const float4 d = *(const float4*)(c11 + o);
    xy[4 * j + 0] = a.x * w00 + b.x * w01 + c.x * w10 + d.x * w11;
    xy[4 * j + 1] = a.y * w00 + b.y * w01 + c.y * w10 + d.y * w11;
    xy[4 * j + 2] = a.z * w00 + b.z * w01 + c.z * w10 + d.z * w11;
    xy[4 * j + 3] = a.w * w00 + b.w * w01 + c.w * w10 + d.w * w11;
  }

  const v8f czero = {};

  // ---- 8 z samples reuse the gathered tile --------------------------------
  for (int zi = 0; zi < NZ; ++zi) {
    const float* zf = zfeat + zi * NF;

    // A operand (16x32 f16): feat = zfeat * xy_feat
    v16h A;
#pragma unroll
    for (int e = 0; e < 16; ++e) {
      const int k = (e < 8) ? (kb + e) : (16 + kb + (e - 8));
      A[e] = (_Float16)(zf[k] * xy[e]);
    }

    // Layer 1: (16x32) @ (32x32) via two 16x16x32 WMMAs
    v8f h1[2];
    h1[0] = wmma_f16(A, B1m[0], czero);
    h1[1] = wmma_f16(A, B1m[1], czero);

    // bias+relu, stage to LDS row-major (m, n) for D->A transpose
#pragma unroll
    for (int h = 0; h < 2; ++h) {
#pragma unroll
      for (int r = 0; r < 8; ++r) {
        const int m = r + 8 * half;         // D layout: vgpr r, lane-half -> M
        const int n = 16 * h + mrow;        // D layout: lane -> N
        hbuf[m * HSTRIDE + n] = fmaxf(h1[h][r] + bias1[h], 0.0f);
      }
    }
    lds_fence();

    // reload in A layout (row = mrow, two contiguous 8-float K runs)
    v16h A2;
    {
      const float* row = hbuf + mrow * HSTRIDE;
      const float4 f0 = *(const float4*)(row + kb);
      const float4 f1 = *(const float4*)(row + kb + 4);
      const float4 f2 = *(const float4*)(row + 16 + kb);
      const float4 f3 = *(const float4*)(row + 16 + kb + 4);
      A2[0] = (_Float16)f0.x; A2[1] = (_Float16)f0.y;
      A2[2] = (_Float16)f0.z; A2[3] = (_Float16)f0.w;
      A2[4] = (_Float16)f1.x; A2[5] = (_Float16)f1.y;
      A2[6] = (_Float16)f1.z; A2[7] = (_Float16)f1.w;
      A2[8]  = (_Float16)f2.x; A2[9]  = (_Float16)f2.y;
      A2[10] = (_Float16)f2.z; A2[11] = (_Float16)f2.w;
      A2[12] = (_Float16)f3.x; A2[13] = (_Float16)f3.y;
      A2[14] = (_Float16)f3.z; A2[15] = (_Float16)f3.w;
    }

    // Layer 2
    v8f h2[2];
    h2[0] = wmma_f16(A2, B2m[0], czero);
    h2[1] = wmma_f16(A2, B2m[1], czero);

#pragma unroll
    for (int h = 0; h < 2; ++h) {
#pragma unroll
      for (int r = 0; r < 8; ++r) {
        const int m = r + 8 * half;
        const int n = 16 * h + mrow;
        hbuf[m * HSTRIDE + n] = fmaxf(h2[h][r] + bias2[h], 0.0f);
      }
    }
    lds_fence();

    v16h A3;
    {
      const float* row = hbuf + mrow * HSTRIDE;
      const float4 f0 = *(const float4*)(row + kb);
      const float4 f1 = *(const float4*)(row + kb + 4);
      const float4 f2 = *(const float4*)(row + 16 + kb);
      const float4 f3 = *(const float4*)(row + 16 + kb + 4);
      A3[0] = (_Float16)f0.x; A3[1] = (_Float16)f0.y;
      A3[2] = (_Float16)f0.z; A3[3] = (_Float16)f0.w;
      A3[4] = (_Float16)f1.x; A3[5] = (_Float16)f1.y;
      A3[6] = (_Float16)f1.z; A3[7] = (_Float16)f1.w;
      A3[8]  = (_Float16)f2.x; A3[9]  = (_Float16)f2.y;
      A3[10] = (_Float16)f2.z; A3[11] = (_Float16)f2.w;
      A3[12] = (_Float16)f3.x; A3[13] = (_Float16)f3.y;
      A3[14] = (_Float16)f3.z; A3[15] = (_Float16)f3.w;
    }

    // Layer 3: B has only column 0 live; lanes with n==0 hold the 16 outputs.
    const v8f d3 = wmma_f16(A3, B3m, czero);

    if (mrow == 0) {
      float* o = out + (size_t)zi * NPIX + tile + 8 * half;  // lane0 -> m 0..7, lane16 -> m 8..15
      float4 o0 = { d3[0] + bias3, d3[1] + bias3, d3[2] + bias3, d3[3] + bias3 };
      float4 o1 = { d3[4] + bias3, d3[5] + bias3, d3[6] + bias3, d3[7] + bias3 };
      *(float4*)(o + 0) = o0;
      *(float4*)(o + 4) = o1;
    }
  }
}

extern "C" void kernel_launch(void* const* d_in, const int* in_sizes, int n_in,
                              void* d_out, int out_size, void* d_ws, size_t ws_size,
                              hipStream_t stream) {
  const float* z    = (const float*)d_in[0];
  const float* data = (const float*)d_in[1];
  const float* zd   = (const float*)d_in[2];
  const float* lw   = (const float*)d_in[3];
  const int*   x0   = (const int*)d_in[4];
  const int*   y0   = (const int*)d_in[5];
  const int*   x1   = (const int*)d_in[6];
  const int*   y1   = (const int*)d_in[7];
  const float* W1   = (const float*)d_in[8];
  const float* b1   = (const float*)d_in[9];
  const float* W2   = (const float*)d_in[10];
  const float* b2   = (const float*)d_in[11];
  const float* W3   = (const float*)d_in[12];
  const float* b3   = (const float*)d_in[13];
  float* out = (float*)d_out;

  // 8 waves/block, 16-pixel tile per wave -> 128 pixels/block
  const int blocks = NPIX / 128;   // 2048
  fused_grid_mlp<<<blocks, 256, 0, stream>>>(z, data, zd, lw, x0, y0, x1, y1,
                                             W1, b1, W2, b2, W3, b3, out);
}
